// GRU_29703993819854
// MI455X (gfx1250) — matrix operand order repaired and checked by reading, not compiled, not run on hardware
//
#include <hip/hip_runtime.h>
#include <math.h>
#include <stdint.h>

typedef __attribute__((ext_vector_type(16))) _Float16 v16h;
typedef __attribute__((ext_vector_type(8)))  float    v8f;

#define HID     64
#define GATES   256      // 4*HID (i,f,g,o)
#define IN_DIM  3
#define OUT_DIM 2
#define PL      300
#define BTILE   16
#define NTHREADS 128     // 4 waves (wave32): one wave per gate block

union Frag16 { v16h v; uint32_t u[8]; _Float16 h[16]; };
union HCvt   { _Float16 h; uint16_t u; };

// ---- branch-free activations on hardware trans ops (v_exp_f32 is base-2) ----
__device__ __forceinline__ float fast_exp2(float x) {
#if __has_builtin(__builtin_amdgcn_exp2f)
    return __builtin_amdgcn_exp2f(x);
#else
    return __expf(x * 0.6931471805599453f);
#endif
}
__device__ __forceinline__ float fast_rcp(float x) {
#if __has_builtin(__builtin_amdgcn_rcpf)
    return __builtin_amdgcn_rcpf(x);
#else
    return 1.0f / x;
#endif
}
__device__ __forceinline__ float fast_sigmoid(float x) {
    return fast_rcp(1.0f + fast_exp2(x * -1.442695040888963f));
}
__device__ __forceinline__ float fast_tanh(float x) {
#if __has_builtin(__builtin_amdgcn_tanhf)
    return __builtin_amdgcn_tanhf(x);
#else
    float e = fast_exp2(x * 2.885390081777927f);
    return 1.0f - 2.0f * fast_rcp(e + 1.0f);
#endif
}

__global__ __launch_bounds__(NTHREADS)
void lstm_seq2seq_kernel(const float* __restrict__ x,
                         const float* __restrict__ force,
                         const float* __restrict__ W_ih,
                         const float* __restrict__ W_hh,
                         const float* __restrict__ b_ih,
                         const float* __restrict__ b_hh,
                         const float* __restrict__ W_out,
                         const float* __restrict__ b_out,
                         float* __restrict__ out,
                         int T)
{
    // ---- LDS: everything on-chip for the whole 811-step recurrence (~56 KB) ----
    __shared__ uint16_t sWhh[GATES * HID];       // f16 bits, row-major [col][k]
    __shared__ float    sWih[GATES * IN_DIM];
    __shared__ float    sBias[GATES];            // b_ih + b_hh
    __shared__ float    sWout[OUT_DIM * HID];
    __shared__ float    sBout[OUT_DIM];
    __shared__ float    sGates[BTILE * GATES];   // pre-activation gates
    __shared__ uint16_t sH[BTILE * HID];         // h state as f16 (A source)
    __shared__ float    sX[BTILE * IN_DIM];      // current step input

    const int tid  = threadIdx.x;
    const int lane = tid & 31;
    const int gate = tid >> 5;            // wave id == gate block (i,f,g,o)
    const int b0   = blockIdx.x * BTILE;  // batch tile base

    // ---- preamble: stage weights, init state ----
    for (int i = tid; i < GATES * HID; i += NTHREADS) {
        HCvt cv; cv.h = (_Float16)W_hh[i];
        sWhh[i] = cv.u;
    }
    for (int i = tid; i < GATES * IN_DIM; i += NTHREADS) sWih[i] = W_ih[i];
    for (int i = tid; i < GATES;          i += NTHREADS) sBias[i] = b_ih[i] + b_hh[i];
    for (int i = tid; i < OUT_DIM * HID;  i += NTHREADS) sWout[i] = W_out[i];
    if (tid < OUT_DIM) sBout[tid] = b_out[tid];
    for (int i = tid; i < BTILE * HID; i += NTHREADS) sH[i] = (uint16_t)0;
    // first step's x
    if (tid < BTILE * IN_DIM) {
        int m = tid / 3, k = tid % 3;
        sX[tid] = x[(size_t)(b0 + m) * T * IN_DIM + k];
    }
    __syncthreads();

    // ---- B fragments (W_hh^T block for this wave's gate) held in VGPRs forever ----
    // B element (k,n) = W_hh[gate*64 + nt*16 + n][k].
    // Layout: lane carries N=lane%16; element s carries K=8h+s (s<8) or 16+8h+(s-8) (s>=8), h=lane/16.
    v16h bf[4][2];
    for (int nt = 0; nt < 4; ++nt)
        for (int kh = 0; kh < 2; ++kh) {
            Frag16 f;
            for (int j = 0; j < 8; ++j) {
                int c0 = gate * 64 + nt * 16 + (lane & 15);
                f.h[j]     = *(const _Float16*)&sWhh[c0 * HID + kh * 32 + (lane >> 4) * 8 + j];
                f.h[j + 8] = *(const _Float16*)&sWhh[c0 * HID + kh * 32 + 16 + (lane >> 4) * 8 + j];
            }
            bf[nt][kh] = f.v;
        }

    // ---- per-lane constants ----
    const int nn = lane & 15;   // N within tile (also A-row M for loads)
    const int hi = lane >> 4;
    float cw0[4], cw1[4], cw2[4], cbb[4]; int ccol[4];
    for (int nt = 0; nt < 4; ++nt) {
        int col = gate * 64 + nt * 16 + nn;
        ccol[nt] = col;
        cw0[nt] = sWih[col * 3 + 0];
        cw1[nt] = sWih[col * 3 + 1];
        cw2[nt] = sWih[col * 3 + 2];
        cbb[nt] = sBias[col];
    }
    // A-fragment word indices (16-bit A 16x32 layout)
    int aw[2][8];
    for (int kh = 0; kh < 2; ++kh)
        for (int j = 0; j < 8; ++j) {
            int k0 = kh * 32 + ((j < 4) ? 0 : 16) + hi * 8 + 2 * (j & 3);
            aw[kh][j] = k0 >> 1;
        }

    // ---- cell state c lives in VGPRs: this thread owns (m = tid>>3, j = (tid&7)*8 + q)
    float creg[8];
    for (int q = 0; q < 8; ++q) creg[q] = 0.f;
    const int em  = tid >> 3;          // elementwise m
    const int ejb = (tid & 7) * 8;     // elementwise j base

    // ---- stage A: gates via WMMA (all 128 threads; EXEC all-ones) ----
    auto wmma_stage = [&]() {
        Frag16 a0, a1;
        const uint32_t* hrow = (const uint32_t*)&sH[nn * HID];
        for (int j = 0; j < 8; ++j) { a0.u[j] = hrow[aw[0][j]]; a1.u[j] = hrow[aw[1][j]]; }

        for (int nt = 0; nt < 4; ++nt) {
            v8f acc;
            for (int p = 0; p < 8; ++p) {      // C init = bias + x @ W_ih^T (K=3 folded)
                int m = p + hi * 8;
                acc[p] = cbb[nt] + sX[m * 3 + 0] * cw0[nt]
                                 + sX[m * 3 + 1] * cw1[nt]
                                 + sX[m * 3 + 2] * cw2[nt];
            }
            acc = __builtin_amdgcn_wmma_f32_16x16x32_f16(false, a0.v, false, bf[nt][0],
                                                         (short)0, acc, false, false);
            acc = __builtin_amdgcn_wmma_f32_16x16x32_f16(false, a1.v, false, bf[nt][1],
                                                         (short)0, acc, false, false);
            for (int p = 0; p < 8; ++p) {
                int m = p + hi * 8;
                sGates[m * GATES + ccol[nt]] = acc[p];
            }
        }
        __syncthreads();
    };

    // ---- stage B: nonlinearity + state update; c stays in VGPRs ----
    auto elem_stage = [&]() {
        for (int q = 0; q < 8; ++q) {
            int j = ejb + q;
            float gi = sGates[em * GATES +       j];
            float gf = sGates[em * GATES +  64 + j];
            float gg = sGates[em * GATES + 128 + j];
            float go = sGates[em * GATES + 192 + j];
            float si = fast_sigmoid(gi);
            float sf = fast_sigmoid(gf);
            float so = fast_sigmoid(go);
            float tg = fast_tanh(gg);
            float c  = sf * creg[q] + si * tg;
            creg[q]  = c;
            float h  = so * fast_tanh(c);
            HCvt cv; cv.h = (_Float16)h;
            sH[em * HID + j] = cv.u;
        }
    };

    // y = h @ W_out^T + b_out ; 4 lanes per (m,o) dot product, shfl reduction
    auto emit_y = [&](int step) {
        int m = tid >> 3;          // 0..15
        int o = (tid >> 2) & 1;    // 0..1
        int part = tid & 3;        // 0..3 : 16 j's each
        float acc = 0.f;
        int jb = part * 16;
        for (int jj = 0; jj < 16; ++jj) {
            int j = jb + jj;
            HCvt cv; cv.u = sH[m * HID + j];
            acc += (float)cv.h * sWout[o * HID + j];
        }
        acc += __shfl_down(acc, 2, 4);
        acc += __shfl_down(acc, 1, 4);
        if (part == 0) {
            acc += sBout[o];
            out[(size_t)(b0 + m) * PL * OUT_DIM + (size_t)step * OUT_DIM + o] = acc;
            sX[m * 3 + o] = acc;
        }
    };

    // ---- encoder: 2 barriers per step ----
    for (int t = 0; t < T; ++t) {
        wmma_stage();                       // reads sX(t), sH(t)  [barrier inside]
        elem_stage();                       // writes sH(t+1)
        if (t + 1 < T && tid < BTILE * IN_DIM) {     // stage x for t+1
            int m = tid / 3, k = tid % 3;
            sX[tid] = x[(size_t)(b0 + m) * T * IN_DIM + (size_t)(t + 1) * IN_DIM + k];
        }
        if (tid < BTILE && t + 32 < T)      // warm lines ~32 steps ahead (counter-free)
            __builtin_prefetch(&x[(size_t)(b0 + tid) * T * IN_DIM + (size_t)(t + 32) * IN_DIM], 0, 0);
        __syncthreads();
    }

    // ---- decoder: y0, then PL-1 autoregressive steps (3 barriers per step) ----
    emit_y(0);                                        // writes sX[.,0..1]
    if (tid >= 64 && tid < 64 + BTILE) {              // force for s=0
        int m = tid - 64;
        sX[m * 3 + 2] = force[(size_t)(b0 + m) * (PL - 1)];
    }
    __syncthreads();

    for (int s = 0; s < PL - 1; ++s) {
        wmma_stage();                                 // reads sX(s), sH  [barrier inside]
        elem_stage();                                 // writes sH
        __syncthreads();
        emit_y(s + 1);                                // reads sH, writes out + sX y-slots
        if (s + 2 < PL && tid >= 64 && tid < 64 + BTILE) {   // force for s+1
            int m = tid - 64;
            sX[m * 3 + 2] = force[(size_t)(b0 + m) * (PL - 1) + (s + 1)];
        }
        __syncthreads();
    }
}

extern "C" void kernel_launch(void* const* d_in, const int* in_sizes, int n_in,
                              void* d_out, int out_size, void* d_ws, size_t ws_size,
                              hipStream_t stream)
{
    const float* x     = (const float*)d_in[0];
    const float* force = (const float*)d_in[1];
    const float* W_ih  = (const float*)d_in[2];
    const float* W_hh  = (const float*)d_in[3];
    const float* b_ih  = (const float*)d_in[4];
    const float* b_hh  = (const float*)d_in[5];
    const float* W_out = (const float*)d_in[6];
    const float* b_out = (const float*)d_in[7];
    float* out = (float*)d_out;

    const int B = in_sizes[1] / (PL - 1);          // force is [B, PL-1, 1] -> 1024
    const int T = in_sizes[0] / (B * IN_DIM);      // x is [B, T, 3]        -> 512

    dim3 grid(B / BTILE), block(NTHREADS);
    hipLaunchKernelGGL(lstm_seq2seq_kernel, grid, block, 0, stream,
                       x, force, W_ih, W_hh, b_ih, b_hh, W_out, b_out, out, T);
}
